// DynamicAdapter_55722905699321
// MI455X (gfx1250) — matrix operand order repaired and checked by reading, not compiled, not run on hardware
//
#include <hip/hip_runtime.h>

// ---------------- WMMA types & tiling ----------------
typedef __attribute__((ext_vector_type(16))) __bf16 v16bf;
typedef __attribute__((ext_vector_type(8)))  float  v8f;

#define BM 128
#define BN 128
#define BK 64
#define LDS_STRIDE 72   // bf16 elems per LDS tile row; 144B = 9*16B keeps b128 alignment

__device__ __forceinline__ float gelu_f(float x) {
    return 0.5f * x * (1.0f + erff(x * 0.70710678118654752f));
}

// ---------------- fragment loads (16-bit A 16x32 / B 32x16, k = 16*half + s in both A and B) ----------------
__device__ __forceinline__ void load_frag_a(const __bf16* As, int r0, int koff, int lane, v16bf& f) {
    const int half = lane >> 4;
    const __bf16* p = As + (r0 + (lane & 15)) * LDS_STRIDE + koff;
    const int kb0 = half * 16;
    const int kb1 = half * 16 + 8;
#pragma unroll
    for (int j = 0; j < 4; ++j) { f[2*j]   = p[kb0 + 2*j]; f[2*j+1]   = p[kb0 + 2*j+1]; }
#pragma unroll
    for (int j = 0; j < 4; ++j) { f[8+2*j] = p[kb1 + 2*j]; f[8+2*j+1] = p[kb1 + 2*j+1]; }
}

__device__ __forceinline__ void load_frag_b(const __bf16* Bs, int c0, int koff, int lane, v16bf& f) {
    const int half = lane >> 4;
    const __bf16* p = Bs + (c0 + (lane & 15)) * LDS_STRIDE + koff + half * 16;
#pragma unroll
    for (int j = 0; j < 16; ++j) f[j] = p[j];
}

__device__ __forceinline__ void mma_step(const __bf16* As, const __bf16* Bs, int koff,
                                         int wm, int wn, int lane, v8f acc[2][4]) {
    v16bf a[2], bfrag[4];
    load_frag_a(As, wm * 32,      koff, lane, a[0]);
    load_frag_a(As, wm * 32 + 16, koff, lane, a[1]);
#pragma unroll
    for (int ni = 0; ni < 4; ++ni) load_frag_b(Bs, wn * 64 + ni * 16, koff, lane, bfrag[ni]);
#pragma unroll
    for (int mi = 0; mi < 2; ++mi)
#pragma unroll
        for (int ni = 0; ni < 4; ++ni)
            acc[mi][ni] = __builtin_amdgcn_wmma_f32_16x16x32_bf16(
                false, a[mi], false, bfrag[ni], (short)0, acc[mi][ni], false, false);
}

// ---------------- async A-tile fill: global -> LDS, 16B chunks, ASYNCcnt tracked ----------------
__device__ __forceinline__ void fill_a_async(__bf16* As, const __bf16* A, int K,
                                             int m0, int k0, int t) {
    const unsigned abase = (unsigned)(size_t)As;   // flat->LDS: low 32 bits are the LDS offset
#pragma unroll
    for (int i = 0; i < 4; ++i) {                  // 128 rows x 64 bf16 = 1024 x 16B chunks
        int c  = t + i * 256;
        int row = c >> 3;
        int cc  = c & 7;
        unsigned lds = abase + row * (LDS_STRIDE * 2) + cc * 16;
        unsigned long long ga =
            (unsigned long long)(size_t)(A + (size_t)(m0 + row) * K + k0 + cc * 8);
        asm volatile("global_load_async_to_lds_b128 %0, %1, off"
                     :: "v"(lds), "v"(ga) : "memory");
    }
}

// ---------------- B-tile register staging + K-transposed b128 LDS store ----------------
// Thread t owns uint column np = t&63 (bf16 cols n=2np, 2np+1) and two K-groups of 8 rows:
// kr = (t>>6)*8 and kr+32. Lo/hi bf16 halves are repacked so K runs contiguously, then
// written as one ds_store_b128 per (row, group): 4 b128 stores/thread vs 32 b16.
__device__ __forceinline__ void load_b_regs(unsigned r[16], const __bf16* B, int N,
                                            int k0, int n0, int t) {
    const unsigned* Bp = (const unsigned*)B;
    const int Nh = N >> 1;
    const int np = t & 63;
#pragma unroll
    for (int g = 0; g < 2; ++g) {
        const int kr = (t >> 6) * 8 + g * 32;
#pragma unroll
        for (int j = 0; j < 8; ++j)
            r[g * 8 + j] = Bp[(size_t)(k0 + kr + j) * Nh + (n0 >> 1) + np];
    }
}

__device__ __forceinline__ void store_b_regs(__bf16* Bs, const unsigned r[16], int t) {
    const int n = (t & 63) * 2;
#pragma unroll
    for (int g = 0; g < 2; ++g) {
        const int kr = (t >> 6) * 8 + g * 32;
        const unsigned* u = r + g * 8;
        uint4 lo, hi;
        lo.x = (u[0] & 0xFFFFu) | (u[1] << 16);
        lo.y = (u[2] & 0xFFFFu) | (u[3] << 16);
        lo.z = (u[4] & 0xFFFFu) | (u[5] << 16);
        lo.w = (u[6] & 0xFFFFu) | (u[7] << 16);
        hi.x = (u[0] >> 16) | (u[1] & 0xFFFF0000u);
        hi.y = (u[2] >> 16) | (u[3] & 0xFFFF0000u);
        hi.z = (u[4] >> 16) | (u[5] & 0xFFFF0000u);
        hi.w = (u[6] >> 16) | (u[7] & 0xFFFF0000u);
        *(uint4*)&Bs[n * LDS_STRIDE + kr]       = lo;   // 144B row stride, kr%8==0 -> 16B aligned
        *(uint4*)&Bs[(n + 1) * LDS_STRIDE + kr] = hi;
    }
}

// ---------------- double-buffered bf16 GEMM mainloop (1 barrier / 64-K step) ----------------
__device__ __forceinline__ void gemm_db_loop(const __bf16* __restrict__ Ag, int K,
                                             const __bf16* __restrict__ Bg, int N,
                                             int m0, int n0, int t,
                                             __bf16* As0, __bf16* As1,
                                             __bf16* Bs0, __bf16* Bs1,
                                             v8f acc[2][4]) {
    const int lane = t & 31, w = t >> 5, wm = w & 3, wn = w >> 2;
    unsigned breg[16];
    fill_a_async(As0, Ag, K, m0, 0, t);
    load_b_regs(breg, Bg, N, 0, n0, t);
    store_b_regs(Bs0, breg, t);
    asm volatile("s_wait_asynccnt 0x0" ::: "memory");
    __syncthreads();
    __bf16 *Ac = As0, *An = As1, *Bc = Bs0, *Bn = Bs1;
    for (int k0 = 0; k0 < K; k0 += BK) {
        const int  k1   = k0 + BK;
        const bool more = (k1 < K);
        if (more) {
            fill_a_async(An, Ag, K, m0, k1, t);    // DMA next A while computing
            load_b_regs(breg, Bg, N, k1, n0, t);   // next B into registers
        }
        mma_step(Ac, Bc, 0,  wm, wn, lane, acc);
        mma_step(Ac, Bc, 32, wm, wn, lane, acc);
        if (more) {
            store_b_regs(Bn, breg, t);
            asm volatile("s_wait_asynccnt 0x0" ::: "memory");
        }
        __syncthreads();
        __bf16* tp = Ac; Ac = An; An = tp;
        tp = Bc; Bc = Bn; Bn = tp;
    }
}

// ---------------- small kernels ----------------
__global__ __launch_bounds__(256) void pool_kernel(const float* __restrict__ X,
                                                   float* __restrict__ pooled) {
    int t = blockIdx.x * 256 + threadIdx.x;        // 4096 = B*H
    int b = t >> 10, h = t & 1023;
    const float* p = X + (size_t)b * 2048 * 1024 + h;
    float s = 0.f;
    for (int si = 0; si < 2048; ++si) s += p[(size_t)si * 1024];
    pooled[t] = s * (1.f / 2048.f);
}

__global__ __launch_bounds__(256) void conv_bf16_kernel(const float* __restrict__ src,
                                                        __bf16* __restrict__ dst, int n) {
    int i = blockIdx.x * 256 + threadIdx.x;
    if (i < n) dst[i] = (__bf16)src[i];
}

__global__ __launch_bounds__(256) void gating_kernel(
    const float* __restrict__ pooled,
    const float* __restrict__ a1w, const float* __restrict__ a1b,
    const float* __restrict__ a2w, const float* __restrict__ a2b,
    const float* __restrict__ a3w, const float* __restrict__ a3b,
    const float* __restrict__ s1w, const float* __restrict__ s1b,
    const float* __restrict__ s2w, const float* __restrict__ s2b,
    const float* __restrict__ s3w, const float* __restrict__ s3b,
    float* __restrict__ gate) {
    __shared__ float buf0[1024];
    __shared__ float buf1[512];
    const int t = threadIdx.x;
    for (int b = 0; b < 4; ++b) {
        for (int i = t; i < 1024; i += 256) buf0[i] = pooled[b * 1024 + i];
        __syncthreads();
        for (int j = t; j < 512; j += 256) {
            float s = a1b[j];
            for (int k = 0; k < 1024; ++k) s += buf0[k] * a1w[k * 512 + j];
            buf1[j] = gelu_f(s);
        }
        __syncthreads();
        if (t < 256) {
            float s = a2b[t];
            for (int k = 0; k < 512; ++k) s += buf1[k] * a2w[k * 256 + t];
            buf0[t] = gelu_f(s);
        }
        __syncthreads();
        if (t < 128) {
            float s = a3b[t];
            for (int k = 0; k < 256; ++k) s += buf0[k] * a3w[k * 128 + t];
            buf1[t] = s;
        }
        __syncthreads();
        if (t < 64) {
            float s = s1b[t];
            for (int k = 0; k < 128; ++k) s += buf1[k] * s1w[k * 64 + t];
            buf0[t] = gelu_f(s);
        }
        __syncthreads();
        if (t < 32) {
            float s = s2b[t];
            for (int k = 0; k < 64; ++k) s += buf0[k] * s2w[k * 32 + t];
            buf1[t] = gelu_f(s);
        }
        __syncthreads();
        if (t < 16) {
            float s = s3b[t];
            for (int k = 0; k < 32; ++k) s += buf1[k] * s3w[k * 16 + t];
            buf0[t] = s;
        }
        __syncthreads();
        if (t == 0) {
            float mx = buf0[0];
            for (int e = 1; e < 16; ++e) mx = fmaxf(mx, buf0[e]);
            float sum = 0.f, ex[16];
            for (int e = 0; e < 16; ++e) { ex[e] = expf(buf0[e] - mx); sum += ex[e]; }
            for (int e = 0; e < 16; ++e) gate[b * 16 + e] = ex[e] / sum;
        }
        __syncthreads();
    }
}

// cgate[b,g] = f_b[g] + sum_e gate[b,e] * (e2_b[e,:] @ f_w[e,:,g])
__global__ __launch_bounds__(256) void cgate_kernel(const float* __restrict__ gate,
                                                    const float* __restrict__ e2b,
                                                    const float* __restrict__ fw,
                                                    const float* __restrict__ fb,
                                                    float* __restrict__ cgate) {
    int t = blockIdx.x * 256 + threadIdx.x;        // 4096 = B*H
    int b = t >> 10, g = t & 1023;
    float acc = fb[g];
    for (int e = 0; e < 16; ++e) {
        const float* fwe = fw + (size_t)e * 1024 * 1024;
        const float* b2e = e2b + e * 1024;
        float s = 0.f;
        for (int h = 0; h < 1024; ++h) s += b2e[h] * fwe[(size_t)h * 1024 + g];
        acc += gate[b * 16 + e] * s;
    }
    cgate[t] = acc;
}

// ---------------- WMMA GEMM kernels ----------------
// W2f[e] = e2_w[e] (512x1024 f32) @ f_w[e] (1024x1024 f32) -> bf16 [512,1024]
__global__ __launch_bounds__(256) void w2f_kernel(const float* __restrict__ e2w,
                                                  const float* __restrict__ fw,
                                                  __bf16* __restrict__ W2f) {
    const int e = blockIdx.z;
    const float* A  = e2w + (size_t)e * 512 * 1024;
    const float* Bm = fw  + (size_t)e * 1024 * 1024;
    __bf16* O = W2f + (size_t)e * 512 * 1024;
    const int m0 = blockIdx.y * BM, n0 = blockIdx.x * BN;
    const int t = threadIdx.x, lane = t & 31, w = t >> 5;
    const int wm = w & 3, wn = w >> 2;
    __shared__ __align__(16) __bf16 As[BM * LDS_STRIDE];
    __shared__ __align__(16) __bf16 Bs[BN * LDS_STRIDE];
    v8f acc[2][4] = {};
    for (int k0 = 0; k0 < 1024; k0 += BK) {
#pragma unroll
        for (int i = 0; i < 32; ++i) {   // A: 128x64 f32 -> bf16
            int idx = t + i * 256, row = idx >> 6, col = idx & 63;
            As[row * LDS_STRIDE + col] = (__bf16)A[(size_t)(m0 + row) * 1024 + k0 + col];
        }
#pragma unroll
        for (int i = 0; i < 32; ++i) {   // B: 64x128 f32 -> bf16, K-transposed
            int idx = t + i * 256, kr = idx >> 7, n = idx & 127;
            Bs[n * LDS_STRIDE + kr] = (__bf16)Bm[(size_t)(k0 + kr) * 1024 + n0 + n];
        }
        __syncthreads();
        mma_step(As, Bs, 0,  wm, wn, lane, acc);
        mma_step(As, Bs, 32, wm, wn, lane, acc);
        __syncthreads();
    }
    const int half = lane >> 4, nl = lane & 15;
#pragma unroll
    for (int mi = 0; mi < 2; ++mi)
#pragma unroll
        for (int ni = 0; ni < 4; ++ni)
#pragma unroll
            for (int v = 0; v < 8; ++v) {
                int row = m0 + wm * 32 + mi * 16 + half * 8 + v;
                int col = n0 + wn * 64 + ni * 16 + nl;
                O[(size_t)row * 1024 + col] = (__bf16)acc[mi][ni][v];
            }
}

// G[row, e*512+col] = gate[b,e] * gelu(Xbf[row,:] @ W1bf[e][:,col] + e1_b[e,col])
__global__ __launch_bounds__(256) void gemm1_kernel(const __bf16* __restrict__ Xbf,
                                                    const __bf16* __restrict__ W1bf,
                                                    const float* __restrict__ e1b,
                                                    const float* __restrict__ gate,
                                                    __bf16* __restrict__ G) {
    const int e = blockIdx.z;
    const int m0 = blockIdx.y * BM, n0 = blockIdx.x * BN;   // n0 in [0,512)
    const int t = threadIdx.x, lane = t & 31, w = t >> 5;
    const int wm = w & 3, wn = w >> 2;
    __shared__ __align__(16) __bf16 As0[BM * LDS_STRIDE], As1[BM * LDS_STRIDE];
    __shared__ __align__(16) __bf16 Bs0[BN * LDS_STRIDE], Bs1[BN * LDS_STRIDE];
    v8f acc[2][4] = {};
    gemm_db_loop(Xbf, 1024, W1bf + (size_t)e * 1024 * 512, 512,
                 m0, n0, t, As0, As1, Bs0, Bs1, acc);
    const float gv = gate[(m0 >> 11) * 16 + e];   // whole 128-row block shares batch b
    const int half = lane >> 4, nl = lane & 15;
#pragma unroll
    for (int mi = 0; mi < 2; ++mi)
#pragma unroll
        for (int ni = 0; ni < 4; ++ni)
#pragma unroll
            for (int v = 0; v < 8; ++v) {
                int row = m0 + wm * 32 + mi * 16 + half * 8 + v;
                int col = n0 + wn * 64 + ni * 16 + nl;
                float x = acc[mi][ni][v] + e1b[e * 512 + col];
                x = gelu_f(x) * gv;
                G[(size_t)row * 8192 + e * 512 + col] = (__bf16)x;
            }
}

// FUSED[row,col] = G[row,:8192] @ W2fstack[:8192,col]   (bf16 x bf16 -> f32)
__global__ __launch_bounds__(256) void gemm2_kernel(const __bf16* __restrict__ G,
                                                    const __bf16* __restrict__ W2f,
                                                    float* __restrict__ FUSED) {
    const int m0 = blockIdx.y * BM, n0 = blockIdx.x * BN;
    const int t = threadIdx.x, lane = t & 31, w = t >> 5;
    const int wm = w & 3, wn = w >> 2;
    __shared__ __align__(16) __bf16 As0[BM * LDS_STRIDE], As1[BM * LDS_STRIDE];
    __shared__ __align__(16) __bf16 Bs0[BN * LDS_STRIDE], Bs1[BN * LDS_STRIDE];
    v8f acc[2][4] = {};
    gemm_db_loop(G, 8192, W2f, 1024, m0, n0, t, As0, As1, Bs0, Bs1, acc);
    const int half = lane >> 4, nl = lane & 15;
#pragma unroll
    for (int mi = 0; mi < 2; ++mi)
#pragma unroll
        for (int ni = 0; ni < 4; ++ni)
#pragma unroll
            for (int v = 0; v < 8; ++v) {
                int row = m0 + wm * 32 + mi * 16 + half * 8 + v;
                int col = n0 + wn * 64 + ni * 16 + nl;
                FUSED[(size_t)row * 1024 + col] = acc[mi][ni][v];
            }
}

// out = LN(X + FUSED + cgate[b]) * ln_g + ln_b
__global__ __launch_bounds__(256) void ln_kernel(const float* __restrict__ X,
                                                 const float* __restrict__ F,
                                                 const float* __restrict__ cg,
                                                 const float* __restrict__ lng,
                                                 const float* __restrict__ lnb,
                                                 float* __restrict__ out) {
    const int row = blockIdx.x;
    const int b = row >> 11;
    const int t = threadIdx.x;
    __shared__ float red[256];
    const float* xr = X + (size_t)row * 1024;
    const float* fr = F + (size_t)row * 1024;
    const float* cr = cg + b * 1024;
    float v[4]; float s = 0.f;
#pragma unroll
    for (int i = 0; i < 4; ++i) {
        int c = t + i * 256;
        v[i] = xr[c] + fr[c] + cr[c];
        s += v[i];
    }
    red[t] = s; __syncthreads();
    for (int st = 128; st > 0; st >>= 1) { if (t < st) red[t] += red[t + st]; __syncthreads(); }
    const float mu = red[0] * (1.f / 1024.f);
    __syncthreads();
    float s2 = 0.f;
#pragma unroll
    for (int i = 0; i < 4; ++i) { float d = v[i] - mu; s2 += d * d; }
    red[t] = s2; __syncthreads();
    for (int st = 128; st > 0; st >>= 1) { if (t < st) red[t] += red[t + st]; __syncthreads(); }
    const float inv = rsqrtf(red[0] * (1.f / 1024.f) + 1e-5f);
    float* orow = out + (size_t)row * 1024;
#pragma unroll
    for (int i = 0; i < 4; ++i) {
        int c = t + i * 256;
        orow[c] = (v[i] - mu) * inv * lng[c] + lnb[c];
    }
}

// ---------------- host launcher ----------------
extern "C" void kernel_launch(void* const* d_in, const int* in_sizes, int n_in,
                              void* d_out, int out_size, void* d_ws, size_t ws_size,
                              hipStream_t stream) {
    (void)in_sizes; (void)n_in; (void)out_size; (void)ws_size;
    const float* X   = (const float*)d_in[0];
    const float* a1w = (const float*)d_in[1];  const float* a1b = (const float*)d_in[2];
    const float* a2w = (const float*)d_in[3];  const float* a2b = (const float*)d_in[4];
    const float* a3w = (const float*)d_in[5];  const float* a3b = (const float*)d_in[6];
    const float* s1w = (const float*)d_in[7];  const float* s1b = (const float*)d_in[8];
    const float* s2w = (const float*)d_in[9];  const float* s2b = (const float*)d_in[10];
    const float* s3w = (const float*)d_in[11]; const float* s3b = (const float*)d_in[12];
    const float* e1w = (const float*)d_in[13]; const float* e1b = (const float*)d_in[14];
    const float* e2w = (const float*)d_in[15]; const float* e2b = (const float*)d_in[16];
    const float* fw  = (const float*)d_in[17]; const float* fb  = (const float*)d_in[18];
    const float* lng = (const float*)d_in[19]; const float* lnb = (const float*)d_in[20];

    char* ws = (char*)d_ws;
    size_t off = 0;
    auto carve = [&](size_t bytes) -> char* {
        char* p = ws + off;
        off += (bytes + 255) & ~(size_t)255;
        return p;
    };
    float*  pooled = (float*)carve(4 * 1024 * sizeof(float));
    float*  gate   = (float*)carve(4 * 16 * sizeof(float));
    float*  cgate  = (float*)carve(4 * 1024 * sizeof(float));
    __bf16* Xbf    = (__bf16*)carve((size_t)8388608 * 2);   // [8192,1024]
    __bf16* W1bf   = (__bf16*)carve((size_t)8388608 * 2);   // [16,1024,512]
    __bf16* W2f    = (__bf16*)carve((size_t)8388608 * 2);   // [16*512,1024]
    __bf16* G      = (__bf16*)carve((size_t)67108864 * 2);  // [8192,8192]
    float*  FUSED  = (float*)carve((size_t)8388608 * 4);    // [8192,1024]

    pool_kernel<<<16, 256, 0, stream>>>(X, pooled);
    gating_kernel<<<1, 256, 0, stream>>>(pooled, a1w, a1b, a2w, a2b, a3w, a3b,
                                         s1w, s1b, s2w, s2b, s3w, s3b, gate);
    conv_bf16_kernel<<<32768, 256, 0, stream>>>(X, Xbf, 8388608);
    conv_bf16_kernel<<<32768, 256, 0, stream>>>(e1w, W1bf, 8388608);
    w2f_kernel<<<dim3(8, 4, 16), 256, 0, stream>>>(e2w, fw, W2f);
    cgate_kernel<<<16, 256, 0, stream>>>(gate, e2b, fw, fb, cgate);
    gemm1_kernel<<<dim3(4, 64, 16), 256, 0, stream>>>(Xbf, W1bf, e1b, gate, G);
    gemm2_kernel<<<dim3(8, 64, 1), 256, 0, stream>>>(G, W2f, FUSED);
    ln_kernel<<<8192, 256, 0, stream>>>(X, FUSED, cgate, lng, lnb, (float*)d_out);
}
